// BiMamba_19215683682663
// MI455X (gfx1250) — matrix operand order repaired and checked by reading, not compiled, not run on hardware
//
#include <hip/hip_runtime.h>

// ---------------------------------------------------------------------------
// BiMamba block for MI455X (gfx1250, wave32, WMMA + Tensor Data Mover).
// GEMMs: bf16 operands, f32 accumulation via v_wmma_f32_16x16x32_bf16,
//        tiles staged global->LDS by TDM (tensor_load_to_lds, TENSORcnt),
//        double-buffered so DMA overlaps the matrix pipe.
// Scan / norms / activations: fp32 VALU.
// ---------------------------------------------------------------------------

#define B_   2
#define L_   2048
#define DM   1024          // d_model
#define DI   2048          // d_inner
#define DS   16            // d_state
#define DC   4             // d_conv
#define DTR  64            // dt_rank
#define NR   (B_ * L_)     // 4096 token rows
#define NXD  (DTR + 2*DS)  // 96
#define EPSV 1e-5f

typedef __bf16 bf16;
typedef __attribute__((ext_vector_type(16))) __bf16       v16bf;
typedef __attribute__((ext_vector_type(8)))  float        v8f;
typedef __attribute__((ext_vector_type(4))) unsigned int  u32x4;
typedef __attribute__((ext_vector_type(8)))  int          i32x8;
typedef __attribute__((ext_vector_type(4)))  int          i32x4;

union FragU { v16bf v; u32x4 q[2]; };

// LDS byte offset of a __shared__ object (addrspace(3) ptrtoint).
__device__ inline unsigned lds_addr_of(const void* p) {
  return (unsigned)(unsigned long long)(__attribute__((address_space(3))) const char*)p;
}

// Issue one TDM 2D tile load: global (row-major, stride0 elems/row) -> LDS
// (tile stored contiguously row-major, tdim1 x tdim0 bf16 elements).
// dim0/dim1 = tensor extents from the tile origin (OOB reads return zero).
__device__ inline void tdm_load_2d(unsigned ldsOff, const bf16* gtile,
                                   unsigned dim0, unsigned dim1,
                                   unsigned stride0,
                                   unsigned tdim0, unsigned tdim1) {
  unsigned long long ga = (unsigned long long)gtile;
  u32x4 g0;
  g0[0] = 1u;                                                  // count=1, is_restore=0
  g0[1] = ldsOff;                                              // lds_addr (bytes)
  g0[2] = (unsigned)(ga & 0xffffffffull);                      // global_addr[31:0]
  g0[3] = (unsigned)((ga >> 32) & 0x01ffffffull) | (2u << 30); // addr[56:32] | type=2
  i32x8 g1;
  g1[0] = (int)(1u << 16);                                     // data_size=1 (2 bytes)
  g1[1] = (int)((dim0 & 0xffffu) << 16);                       // tensor_dim0[15:0]
  g1[2] = (int)((dim0 >> 16) | ((dim1 & 0xffffu) << 16));      // dim0[31:16], dim1[15:0]
  g1[3] = (int)((dim1 >> 16) | (tdim0 << 16));                 // dim1[31:16], tile_dim0
  g1[4] = (int)(tdim1 & 0xffffu);                              // tile_dim1 (tile_dim2=0)
  g1[5] = (int)stride0;                                        // tensor_dim0_stride[31:0]
  g1[6] = 0;                                                   // stride0[47:32], dim1_stride lo
  g1[7] = 0;
  i32x4 z4 = {0, 0, 0, 0};
  i32x8 z8 = {0, 0, 0, 0, 0, 0, 0, 0};
  __builtin_amdgcn_tensor_load_to_lds(g0, g1, z4, z4, z8, 0);  // 6-arg toolchain variant
}

// ------------------------- elementwise helpers -----------------------------

__global__ void cast_bf16_flat(const float* __restrict__ in, bf16* __restrict__ out, int n) {
  int i = blockIdx.x * blockDim.x + threadIdx.x;
  if (i < n) out[i] = (bf16)in[i];
}

// Transpose-cast: in (K x N, f32, row-major) -> out (N x K, bf16, row-major).
__global__ void cast_transpose_bf16(const float* __restrict__ in, bf16* __restrict__ out,
                                    int K, int N) {
  int i = blockIdx.x * blockDim.x + threadIdx.x;
  if (i >= K * N) return;
  int k = i / N, n = i % N;
  out[(size_t)n * K + k] = (bf16)in[i];
}

// x (B,L,DM) fp32 -> bf16, with optional flip along L (for backward direction)
__global__ void cast_bf16_flip(const float* __restrict__ in, bf16* __restrict__ out, int flip) {
  int i = blockIdx.x * blockDim.x + threadIdx.x;
  if (i >= NR * DM) return;
  int r = i / DM, c = i % DM;
  int b = r / L_, t = r % L_;
  int sr = b * L_ + (flip ? (L_ - 1 - t) : t);
  out[(size_t)r * DM + c] = (bf16)in[(size_t)sr * DM + c];
}

// ------------------------------- WMMA GEMM ---------------------------------
// C[M,N](f32) [and Cb bf16] = act(A[M,K](bf16) @ B[K,N](bf16) + bias)
// B operand is passed PRE-TRANSPOSED: Bt is N x K row-major (ldbt = row stride).
// Block tile 128x128, BK=64, TDM-staged, double-buffered LDS.
#define BM  128
#define BN  128
#define BKK 64

__launch_bounds__(256)
__global__ void gemm_bf16_tdm_kernel(const bf16* __restrict__ A, int lda,
                                     const bf16* __restrict__ Bt, int ldbt,
                                     float* __restrict__ C, bf16* __restrict__ Cb, int ldc,
                                     const float* __restrict__ bias,
                                     int M, int N, int K, int act) {
  __shared__ bf16 sA[2][BM * BKK];        // [m][k] tile, row-major
  __shared__ bf16 sB[2][BN * BKK];        // [n][k] tile, row-major (from Bt)

  int tid   = threadIdx.x;
  int lane  = tid & 31;
  int wave  = tid >> 5;
  int waveM = wave & 3;                   // 4 waves along M (32 rows each)
  int waveN = wave >> 2;                  // 2 waves along N (64 cols each)
  int mBase = blockIdx.y * BM;
  int nBase = blockIdx.x * BN;
  int r     = lane & 15;
  int half  = lane >> 4;
  bool issuer = (wave == 0);

  const bf16* Atile = A  + (size_t)mBase * lda;
  const bf16* Btile = Bt + (size_t)nBase * ldbt;
  unsigned dimA1 = (unsigned)(M - mBase); // rows remaining from tile origin
  unsigned dimB1 = (unsigned)(N - nBase);

  v8f acc[2][4];
#pragma unroll
  for (int i = 0; i < 2; ++i)
#pragma unroll
    for (int j = 0; j < 4; ++j) acc[i][j] = 0;

  // prologue: DMA first K-tile into buffer 0
  if (issuer) {
    tdm_load_2d(lds_addr_of(sA[0]), Atile, (unsigned)lda,  dimA1, (unsigned)lda,  BKK, BM);
    tdm_load_2d(lds_addr_of(sB[0]), Btile, (unsigned)ldbt, dimB1, (unsigned)ldbt, BKK, BN);
  }

  int nk = K / BKK;
  for (int ik = 0; ik < nk; ++ik) {
    int p = ik & 1;
    __syncthreads();                       // everyone done with buffer p^1
    if (issuer) {
      if (ik + 1 < nk) {
        int k1 = (ik + 1) * BKK;
        tdm_load_2d(lds_addr_of(sA[p ^ 1]), Atile + k1, (unsigned)lda,  dimA1,
                    (unsigned)lda,  BKK, BM);
        tdm_load_2d(lds_addr_of(sB[p ^ 1]), Btile + k1, (unsigned)ldbt, dimB1,
                    (unsigned)ldbt, BKK, BN);
        __builtin_amdgcn_s_wait_tensorcnt(2);   // buffer p's two loads complete
      } else {
        __builtin_amdgcn_s_wait_tensorcnt(0);
      }
    }
    __syncthreads();                       // publish buffer p to all waves

#pragma unroll
    for (int kk = 0; kk < BKK; kk += 32) {
      FragU a[2], b[4];
      // A fragment per ISA 16-bit A 16x32 layout:
      // lanes 0-15: K 0..7 then 16..23 ; lanes 16-31: K 8..15 then 24..31
#pragma unroll
      for (int i = 0; i < 2; ++i) {
        const bf16* pa = &sA[p][(waveM * 32 + i * 16 + r) * BKK + kk + half * 8];
        a[i].q[0] = *(const u32x4*)pa;
        a[i].q[1] = *(const u32x4*)(pa + 16);
      }
      // B fragment (same K order as A): lanes 0-15: K 0..7 then 16..23 ; lanes 16-31: K 8..15 then 24..31
#pragma unroll
      for (int j = 0; j < 4; ++j) {
        const bf16* pb = &sB[p][(waveN * 64 + j * 16 + r) * BKK + kk + half * 8];
        b[j].q[0] = *(const u32x4*)pb;
        b[j].q[1] = *(const u32x4*)(pb + 16);
      }
#pragma unroll
      for (int i = 0; i < 2; ++i)
#pragma unroll
        for (int j = 0; j < 4; ++j)
          acc[i][j] = __builtin_amdgcn_wmma_f32_16x16x32_bf16(
              false, a[i].v, false, b[j].v, (short)0, acc[i][j], false, false);
    }
  }

  // ---- epilogue: bias + activation + store (f32 C/D layout: VGPR e -> M = half*8+e) ----
#pragma unroll
  for (int j = 0; j < 4; ++j) {
    int col = nBase + waveN * 64 + j * 16 + r;
    if (col >= N) continue;
    float bv = bias ? bias[col] : 0.0f;
#pragma unroll
    for (int i = 0; i < 2; ++i) {
#pragma unroll
      for (int e = 0; e < 8; ++e) {
        int row = mBase + waveM * 32 + i * 16 + half * 8 + e;
        float v = acc[i][j][e] + bv;
        if (act == 1)      v = v / (1.0f + __expf(-v));                      // silu
        else if (act == 2) v = fmaxf(v, 0.0f);                               // relu
        else if (act == 3) v = (v > 20.0f) ? v : __logf(1.0f + __expf(v));   // softplus
        if (C)  C [(size_t)row * ldc + col] = v;
        if (Cb) Cb[(size_t)row * ldc + col] = (bf16)v;
      }
    }
  }
}

// ------------------- depthwise causal conv (DC=4) + silu -------------------
// xz is (NR, 2*DI); xi occupies cols [0,DI)
__global__ void dwconv_silu_kernel(const float* __restrict__ xz,
                                   const float* __restrict__ w,
                                   const float* __restrict__ bconv,
                                   float* __restrict__ of, bf16* __restrict__ ob) {
  int i = blockIdx.x * blockDim.x + threadIdx.x;
  if (i >= NR * DI) return;
  int rrow = i / DI, d = i % DI;
  int t = rrow % L_;
  float acc = bconv[d];
#pragma unroll
  for (int k = 0; k < DC; ++k) {
    int tt = t - (DC - 1) + k;
    if (tt >= 0) acc += xz[(size_t)(rrow - (DC - 1) + k) * (2 * DI) + d] * w[d * DC + k];
  }
  float s = acc / (1.0f + __expf(-acc));
  of[i] = s;
  ob[i] = (bf16)s;
}

// ----------------------------- selective scan ------------------------------
// One thread per (b, d_inner) channel; h[16] in registers; B/C via LDS.
__launch_bounds__(256)
__global__ void scan_kernel(const float* __restrict__ u,     // NR x DI
                            const float* __restrict__ dt,    // NR x DI
                            const float* __restrict__ xdbc,  // NR x 96 (B@64, C@80)
                            const float* __restrict__ Alog,  // DI x DS
                            const float* __restrict__ Dp,    // DI
                            float* __restrict__ y) {         // NR x DI
  int d = blockIdx.x * blockDim.x + threadIdx.x;
  int b = blockIdx.y;
  float Arow[DS];
#pragma unroll
  for (int s = 0; s < DS; ++s) Arow[s] = -__expf(Alog[(size_t)d * DS + s]);
  float Dv = Dp[d];
  float h[DS];
#pragma unroll
  for (int s = 0; s < DS; ++s) h[s] = 0.0f;

  __shared__ float sBC[2 * DS];
  for (int t = 0; t < L_; ++t) {
    int row = b * L_ + t;
    if (threadIdx.x < 2 * DS)
      sBC[threadIdx.x] = xdbc[(size_t)row * NXD + DTR + threadIdx.x];
    __syncthreads();
    float dtv = dt[(size_t)row * DI + d];
    float uv  = u [(size_t)row * DI + d];
    float du  = dtv * uv;
    float acc = 0.0f;
#pragma unroll
    for (int s = 0; s < DS; ++s) {
      float dA = __expf(dtv * Arow[s]);
      h[s] = dA * h[s] + du * sBC[s];
      acc += h[s] * sBC[DS + s];
    }
    y[(size_t)row * DI + d] = acc + uv * Dv;
    __syncthreads();
  }
}

// ------------------------------ gating: y*silu(z) --------------------------
__global__ void gate_kernel(const float* __restrict__ ys, const float* __restrict__ xz,
                            bf16* __restrict__ g) {
  int i = blockIdx.x * blockDim.x + threadIdx.x;
  if (i >= NR * DI) return;
  int rrow = i / DI, d = i % DI;
  float z = xz[(size_t)rrow * (2 * DI) + DI + d];
  float s = z / (1.0f + __expf(-z));
  g[i] = (bf16)(ys[i] * s);
}

// ------------------- residual add (+optional flip) + layernorm -------------
__launch_bounds__(256)
__global__ void add_ln_kernel(const float* __restrict__ yin, int flip,
                              const float* __restrict__ resid,
                              const float* __restrict__ w, const float* __restrict__ bb,
                              float* __restrict__ out) {
  __shared__ float red[256];
  int rrow = blockIdx.x;
  int b = rrow / L_, t = rrow % L_;
  int yr = flip ? (b * L_ + (L_ - 1 - t)) : rrow;
  int tid = threadIdx.x;
  float loc[4];
  float s = 0.0f;
#pragma unroll
  for (int i = 0; i < 4; ++i) {
    int c = tid + i * 256;
    loc[i] = yin[(size_t)yr * DM + c] + resid[(size_t)rrow * DM + c];
    s += loc[i];
  }
  red[tid] = s; __syncthreads();
  for (int off = 128; off > 0; off >>= 1) {
    if (tid < off) red[tid] += red[tid + off];
    __syncthreads();
  }
  float mu = red[0] / DM;
  __syncthreads();
  float vs = 0.0f;
#pragma unroll
  for (int i = 0; i < 4; ++i) { float dl = loc[i] - mu; vs += dl * dl; }
  red[tid] = vs; __syncthreads();
  for (int off = 128; off > 0; off >>= 1) {
    if (tid < off) red[tid] += red[tid + off];
    __syncthreads();
  }
  float rs = rsqrtf(red[0] / DM + EPSV);
#pragma unroll
  for (int i = 0; i < 4; ++i) {
    int c = tid + i * 256;
    out[(size_t)rrow * DM + c] = (loc[i] - mu) * rs * w[c] + bb[c];
  }
}

// ----------------------------- yf + yb (+bf16) -----------------------------
__global__ void sum_kernel(const float* __restrict__ a, const float* __restrict__ b,
                           float* __restrict__ o, bf16* __restrict__ ob) {
  int i = blockIdx.x * blockDim.x + threadIdx.x;
  if (i >= NR * DM) return;
  float v = a[i] + b[i];
  o[i] = v;
  ob[i] = (bf16)v;
}

// --------------------------------- driver ----------------------------------

extern "C" void kernel_launch(void* const* d_in, const int* in_sizes, int n_in,
                              void* d_out, int out_size, void* d_ws, size_t ws_size,
                              hipStream_t stream) {
  (void)in_sizes; (void)n_in; (void)out_size; (void)ws_size;
  const float* x        = (const float*)d_in[0];
  const float* Win[2]   = {(const float*)d_in[1],  (const float*)d_in[10]};
  const float* convw[2] = {(const float*)d_in[2],  (const float*)d_in[11]};
  const float* convb[2] = {(const float*)d_in[3],  (const float*)d_in[12]};
  const float* Wx[2]    = {(const float*)d_in[4],  (const float*)d_in[13]};
  const float* Wdt[2]   = {(const float*)d_in[5],  (const float*)d_in[14]};
  const float* bdt[2]   = {(const float*)d_in[6],  (const float*)d_in[15]};
  const float* Alog[2]  = {(const float*)d_in[7],  (const float*)d_in[16]};
  const float* Dp[2]    = {(const float*)d_in[8],  (const float*)d_in[17]};
  const float* Wout[2]  = {(const float*)d_in[9],  (const float*)d_in[18]};
  const float* fwn_w = (const float*)d_in[19];
  const float* fwn_b = (const float*)d_in[20];
  const float* fin_w = (const float*)d_in[21];
  const float* fin_b = (const float*)d_in[22];
  const float* ffW1  = (const float*)d_in[23];
  const float* ffb1  = (const float*)d_in[24];
  const float* ffW2  = (const float*)d_in[25];
  const float* ffb2  = (const float*)d_in[26];
  float* out = (float*)d_out;

  char* base = (char*)d_ws;
  size_t off = 0;
  auto carve = [&](size_t bytes) -> char* {
    char* p = base + off;
    off += (bytes + 255) & ~(size_t)255;
    return p;
  };

  // bf16 TRANSPOSED weight copies (N x K row-major for the TDM B-operand path)
  bf16* wWin[2], *wWx[2], *wWdt[2], *wWout[2];
  for (int d = 0; d < 2; ++d) {
    wWin[d]  = (bf16*)carve((size_t)DM * 2 * DI * 2);
    wWx[d]   = (bf16*)carve((size_t)DI * NXD * 2);
    wWdt[d]  = (bf16*)carve((size_t)DTR * DI * 2);
    wWout[d] = (bf16*)carve((size_t)DI * DM * 2);
  }
  bf16* wW1 = (bf16*)carve((size_t)DM * 2 * DM * 2);
  bf16* wW2 = (bf16*)carve((size_t)2 * DM * DM * 2);

  // per-direction scratch (reused sequentially for fw then bw)
  bf16*  x_bf     = (bf16*)carve((size_t)NR * DM * 2);
  float* xz_f     = (float*)carve((size_t)NR * 2 * DI * 4);
  float* xic_f    = (float*)carve((size_t)NR * DI * 4);
  bf16*  xic_b    = (bf16*)carve((size_t)NR * DI * 2);
  float* xdbc_f   = (float*)carve((size_t)NR * NXD * 4);
  bf16*  xdbc_b   = (bf16*)carve((size_t)NR * NXD * 2);
  float* dt_f     = (float*)carve((size_t)NR * DI * 4);
  float* yscan_f  = (float*)carve((size_t)NR * DI * 4);
  bf16*  g_b      = (bf16*)carve((size_t)NR * DI * 2);
  float* ydir_f   = (float*)carve((size_t)NR * DM * 4);
  // persistent across directions / post phase
  float* yf_ln    = (float*)carve((size_t)NR * DM * 4);
  float* yb_ln    = (float*)carve((size_t)NR * DM * 4);
  float* ysum_f   = (float*)carve((size_t)NR * DM * 4);
  bf16*  ysum_b   = (bf16*)carve((size_t)NR * DM * 2);
  bf16*  ff1_b    = (bf16*)carve((size_t)NR * 2 * DM * 2);
  float* ff2_f    = (float*)carve((size_t)NR * DM * 4);

  auto castwT = [&](const float* src, bf16* dst, int K, int N) {
    int n = K * N;
    cast_transpose_bf16<<<(n + 255) / 256, 256, 0, stream>>>(src, dst, K, N);
  };
  // A: M x K (lda), Bt: N x K (ldbt = K of that GEMM)
  auto gemm = [&](const bf16* A, int lda, const bf16* Bt, int ldbt,
                  float* C, bf16* Cb, int ldc, const float* bias,
                  int M, int N, int K, int act) {
    dim3 grid((N + BN - 1) / BN, (M + BM - 1) / BM);
    gemm_bf16_tdm_kernel<<<grid, 256, 0, stream>>>(A, lda, Bt, ldbt, C, Cb, ldc,
                                                   bias, M, N, K, act);
  };

  // --- weight conversion + transpose (once per call) ---
  for (int d = 0; d < 2; ++d) {
    castwT(Win[d],  wWin[d],  DM,  2 * DI);
    castwT(Wx[d],   wWx[d],   DI,  NXD);
    castwT(Wdt[d],  wWdt[d],  DTR, DI);
    castwT(Wout[d], wWout[d], DI,  DM);
  }
  castwT(ffW1, wW1, DM, 2 * DM);
  castwT(ffW2, wW2, 2 * DM, DM);

  const int nEl2 = NR * DI;   // 8.4M
  const int nEl1 = NR * DM;   // 4.2M

  // --- two mamba directions ---
  for (int d = 0; d < 2; ++d) {
    int flip = (d == 1);
    cast_bf16_flip<<<(nEl1 + 255) / 256, 256, 0, stream>>>(x, x_bf, flip);

    // xz = x @ Win                       (4096 x 4096, K=1024)
    gemm(x_bf, DM, wWin[d], DM, xz_f, nullptr, 2 * DI, nullptr, NR, 2 * DI, DM, 0);
    // xi = silu(causal dwconv(xi))
    dwconv_silu_kernel<<<(nEl2 + 255) / 256, 256, 0, stream>>>(xz_f, convw[d], convb[d], xic_f, xic_b);
    // xdbc = xi @ Wx                     (4096 x 96,  K=2048)
    gemm(xic_b, DI, wWx[d], DI, xdbc_f, xdbc_b, NXD, nullptr, NR, NXD, DI, 0);
    // dt = softplus(dtin @ Wdt + bdt)    (4096 x 2048, K=64)
    gemm(xdbc_b, NXD, wWdt[d], DTR, dt_f, nullptr, DI, bdt[d], NR, DI, DTR, 3);
    // selective scan
    scan_kernel<<<dim3(DI / 256, B_), 256, 0, stream>>>(xic_f, dt_f, xdbc_f, Alog[d], Dp[d], yscan_f);
    // g = bf16(y * silu(z))
    gate_kernel<<<(nEl2 + 255) / 256, 256, 0, stream>>>(yscan_f, xz_f, g_b);
    // ydir = g @ Wout                    (4096 x 1024, K=2048)
    gemm(g_b, DI, wWout[d], DI, ydir_f, nullptr, DM, nullptr, NR, DM, DI, 0);
    // LN(flip?(ydir) + x) with fwn params (reference uses fwn for both dirs)
    add_ln_kernel<<<NR, 256, 0, stream>>>(ydir_f, flip, x, fwn_w, fwn_b, flip ? yb_ln : yf_ln);
  }

  // --- feed-forward + final layernorm ---
  sum_kernel<<<(nEl1 + 255) / 256, 256, 0, stream>>>(yf_ln, yb_ln, ysum_f, ysum_b);
  // ff1 = relu(y @ W1 + b1)              (4096 x 2048, K=1024)
  gemm(ysum_b, DM, wW1, DM, nullptr, ff1_b, 2 * DM, ffb1, NR, 2 * DM, DM, 2);
  // ff2 = ff1 @ W2 + b2                  (4096 x 1024, K=2048)
  gemm(ff1_b, 2 * DM, wW2, 2 * DM, ff2_f, nullptr, DM, ffb2, NR, DM, 2 * DM, 0);
  // out = LN(ff2 + y)
  add_ln_kernel<<<NR, 256, 0, stream>>>(ff2_f, 0, ysum_f, fin_w, fin_b, out);
}
